// OwnMultiHeadTransformer_55654186221770
// MI455X (gfx1250) — compile-verified, matrix operand order checked by reading, not run on hardware
//
#include <hip/hip_runtime.h>
#include <hip/hip_bf16.h>

// ---------------------------------------------------------------------------
// Transformer block for MI455X (gfx1250, wave32, WMMA f16 -> f32 accum)
//   B=4, S=2048, D=1024, H=16, HD=64, FF=4096, M = B*S = 8192
// Pipeline:
//   1) f32 -> f16 conversion of embeddings + all weights
//   2) WMMA GEMM (async-LDS double-buffered): Q/K/V = emb @ W^T + b
//   3) Flash-attention with WMMA (QK^T and P@V), online softmax
//   4) WMMA GEMM + ReLU: hid = Hcat @ W1^T + b1
//   5) WMMA GEMM (f32 out): out = hid @ W2^T + b2 -> d_out
// CDNA5 specifics: v_wmma_f32_16x16x32_f16, global_load_async_to_lds_b128
// (ASYNCcnt pipeline), s_wait_asynccnt / s_wait_dscnt, wave32 shuffles.
// ---------------------------------------------------------------------------

typedef __attribute__((ext_vector_type(16))) _Float16 v16h;
typedef __attribute__((ext_vector_type(8)))  float    v8f;

#define TB 4
#define TS 2048
#define TD 1024
#define TH 16
#define THD 64
#define TFF 4096
#define TM (TB * TS)   // 8192

// async copy of 16 bytes (8 halves) global -> LDS, tracked by ASYNCcnt
__device__ __forceinline__ void async_copy_b128(unsigned lds_byte_addr,
                                                const _Float16* gptr) {
    asm volatile("global_load_async_to_lds_b128 %0, %1, off"
                 :: "v"(lds_byte_addr), "v"(gptr)
                 : "memory");
}
__device__ __forceinline__ void wait_async(int n) {
    if (n == 0) asm volatile("s_wait_asynccnt 0x0" ::: "memory");
    else        asm volatile("s_wait_asynccnt 0x4" ::: "memory");
}

// -------------------------------- f32 -> f16 -------------------------------
__global__ __launch_bounds__(256) void cvt_f32_to_f16(const float* __restrict__ in,
                                                      _Float16* __restrict__ out,
                                                      int n) {
    int i = blockIdx.x * blockDim.x + threadIdx.x;
    if (i < n) out[i] = (_Float16)in[i];
}

// ------------------------- WMMA tiled GEMM (C = A @ B^T) -------------------
// A: [M,K] f16 row-major. Bw: [N,K] f16 row-major (torch Linear weight).
// bias: f32 [N] or null. C: [M,N] OutT row-major. Optional ReLU.
// Block tile 128x128, K-step 32, 256 threads = 8 waves, wave tile 32x64.
// Double-buffered LDS fed by global_load_async_to_lds_b128.
template <bool RELU, typename OutT>
__global__ __launch_bounds__(256) void gemm_wmma(const _Float16* __restrict__ A,
                                                 const _Float16* __restrict__ Bw,
                                                 const float* __restrict__ bias,
                                                 OutT* __restrict__ C,
                                                 int M, int N, int K) {
    __shared__ _Float16 As[2][128 * 32];
    __shared__ _Float16 Bs[2][128 * 32];

    const int tid  = threadIdx.x;
    const int wave = tid >> 5;
    const int lane = tid & 31;
    const int wm   = wave >> 1;     // 0..3 : wave row
    const int wn   = wave & 1;      // 0..1 : wave col
    const int r    = lane & 15;
    const int hs   = lane >> 4;     // 0/1, selects K-half per ISA A/B layout

    const int blockM = blockIdx.y * 128;
    const int blockN = blockIdx.x * 128;

    const v8f vzero = {0.f, 0.f, 0.f, 0.f, 0.f, 0.f, 0.f, 0.f};
    v8f acc[2][4];
#pragma unroll
    for (int i = 0; i < 2; ++i)
#pragma unroll
        for (int j = 0; j < 4; ++j) acc[i][j] = vzero;

    // cooperative tile loads: thread t owns 16 halves of A and of B per K-step
    const int ldr = tid >> 1;          // 0..127
    const int ldc = (tid & 1) * 16;    // 0 or 16

    const unsigned ldsA[2] = {
        (unsigned)(uintptr_t)(&As[0][ldr * 32 + ldc]),
        (unsigned)(uintptr_t)(&As[1][ldr * 32 + ldc])};
    const unsigned ldsB[2] = {
        (unsigned)(uintptr_t)(&Bs[0][ldr * 32 + ldc]),
        (unsigned)(uintptr_t)(&Bs[1][ldr * 32 + ldc])};
    const _Float16* gA = A  + (size_t)(blockM + ldr) * K + ldc;
    const _Float16* gB = Bw + (size_t)(blockN + ldr) * K + ldc;

    auto issue = [&](int k0, int buf) {
        async_copy_b128(ldsA[buf],      gA + k0);
        async_copy_b128(ldsA[buf] + 16, gA + k0 + 8);
        async_copy_b128(ldsB[buf],      gB + k0);
        async_copy_b128(ldsB[buf] + 16, gB + k0 + 8);
    };

    issue(0, 0);                              // prologue: tile 0 in flight
    for (int k0 = 0; k0 < K; k0 += 32) {
        const int buf = (k0 >> 5) & 1;
        if (k0 + 32 < K) { issue(k0 + 32, buf ^ 1); wait_async(4); }
        else             { wait_async(0); }
        __syncthreads();                      // tile `buf` resident for all

        // A fragment per ISA 16-bit 16x32 layout:
        //   lane hs half: halves 0..7 at k = hs*8 .. hs*8+7, halves 8..15 at +16
        v16h af[2], bf[4];
#pragma unroll
        for (int i = 0; i < 2; ++i) {
            const _Float16* ab = &As[buf][(wm * 32 + i * 16 + r) * 32 + hs * 8];
            ((uint4*)&af[i])[0] = *(const uint4*)(ab);
            ((uint4*)&af[i])[1] = *(const uint4*)(ab + 16);
        }
#pragma unroll
        for (int j = 0; j < 4; ++j) {
            const _Float16* bb = &Bs[buf][(wn * 64 + j * 16 + r) * 32 + hs * 8];
            ((uint4*)&bf[j])[0] = *(const uint4*)(bb);
            ((uint4*)&bf[j])[1] = *(const uint4*)(bb + 16);
        }
#pragma unroll
        for (int i = 0; i < 2; ++i)
#pragma unroll
            for (int j = 0; j < 4; ++j)
                acc[i][j] = __builtin_amdgcn_wmma_f32_16x16x32_f16(
                    false, af[i], false, bf[j], (short)0, acc[i][j], false, false);

        __syncthreads();                      // everyone done before overwrite
    }

    // epilogue: C/D layout = lane (hs,r): col n = r, rows = hs*8 + 0..7
#pragma unroll
    for (int j = 0; j < 4; ++j) {
        const int n   = blockN + wn * 64 + j * 16 + r;
        const float bv = bias ? bias[n] : 0.f;
#pragma unroll
        for (int i = 0; i < 2; ++i) {
#pragma unroll
            for (int jj = 0; jj < 8; ++jj) {
                const int m = blockM + wm * 32 + i * 16 + hs * 8 + jj;
                float v = acc[i][j][jj] + bv;
                if (RELU) v = fmaxf(v, 0.f);
                C[(size_t)m * N + n] = (OutT)v;
            }
        }
    }
}

// ------------------------------ flash attention ----------------------------
// grid = (S/64, B*H), block = 128 (4 waves); each wave owns 16 query rows.
// Q/K/V in [B,S,H,HD] f16 layout (row stride D). Output Hcat same layout.
__global__ __launch_bounds__(128) void attn_wmma(const _Float16* __restrict__ Qg,
                                                 const _Float16* __restrict__ Kg,
                                                 const _Float16* __restrict__ Vg,
                                                 _Float16* __restrict__ Og) {
    constexpr int LDT = 72;                 // padded LDS row stride (halves)
    __shared__ _Float16 Qs[64 * LDT];
    __shared__ _Float16 Ks[64 * LDT];
    __shared__ _Float16 Vt[64 * LDT];       // V transposed: [e][k]
    __shared__ _Float16 Ps[64 * LDT];       // 4 waves x 16 rows of P

    const int bh   = blockIdx.y;            // 0..63
    const int b    = bh >> 4;
    const int h    = bh & 15;
    const int q0   = blockIdx.x * 64;
    const int tid  = threadIdx.x;
    const int wave = tid >> 5;
    const int lane = tid & 31;
    const int r    = lane & 15;
    const int hs   = lane >> 4;

    const size_t rowbase = (size_t)(b * TS) * TD + (size_t)h * THD;
    const _Float16* Qb = Qg + rowbase;
    const _Float16* Kb = Kg + rowbase;
    const _Float16* Vb = Vg + rowbase;

    // load Q block: 64 rows x 64 cols (async global -> LDS)
#pragma unroll
    for (int c = 0; c < 4; ++c) {
        int idx = (c * 128 + tid) * 8;      // 0..4095 in steps of 8
        int row = idx >> 6;
        int col = idx & 63;
        async_copy_b128((unsigned)(uintptr_t)(Qs + row * LDT + col),
                        Qb + (size_t)(q0 + row) * TD + col);
    }

    const v8f vzero = {0.f, 0.f, 0.f, 0.f, 0.f, 0.f, 0.f, 0.f};
    v8f o[4];
#pragma unroll
    for (int j = 0; j < 4; ++j) o[j] = vzero;
    float mrow[8], lrow[8];
#pragma unroll
    for (int jj = 0; jj < 8; ++jj) { mrow[jj] = -1.0e30f; lrow[jj] = 0.f; }

    const float scale = 0.03125f;           // 1/sqrt(D) = 1/32

    for (int kt = 0; kt < TS / 64; ++kt) {
        const int k0 = kt * 64;
        __syncthreads();                    // previous tile fully consumed
        // K tile via async-to-LDS; V tile transposed via VGPR scatter
#pragma unroll
        for (int c = 0; c < 4; ++c) {
            int idx = (c * 128 + tid) * 8;
            int row = idx >> 6;
            int col = idx & 63;
            async_copy_b128((unsigned)(uintptr_t)(Ks + row * LDT + col),
                            Kb + (size_t)(k0 + row) * TD + col);
            uint4 vv = *(const uint4*)(Vb + (size_t)(k0 + row) * TD + col);
            const _Float16* vp = (const _Float16*)&vv;
#pragma unroll
            for (int u = 0; u < 8; ++u) Vt[(col + u) * LDT + row] = vp[u];
        }
        if (kt + 1 < TS / 64)               // lowers to global_prefetch_b8
            __builtin_prefetch(Kb + (size_t)(k0 + 64) * TD + (tid & 63), 0, 1);
        wait_async(0);                      // own async K (and Q) writes done
        __syncthreads();

        // ---- scores = Q @ K^T  (16 rows x 64 keys per wave) ----
        v8f sc[4];
#pragma unroll
        for (int j = 0; j < 4; ++j) sc[j] = vzero;
#pragma unroll
        for (int ks = 0; ks < 2; ++ks) {
            v16h aq;
            const _Float16* ab = Qs + (wave * 16 + r) * LDT + ks * 32 + hs * 8;
            ((uint4*)&aq)[0] = *(const uint4*)(ab);
            ((uint4*)&aq)[1] = *(const uint4*)(ab + 16);
#pragma unroll
            for (int j = 0; j < 4; ++j) {
                v16h bk;
                const _Float16* bb = Ks + (j * 16 + r) * LDT + ks * 32 + hs * 8;
                ((uint4*)&bk)[0] = *(const uint4*)(bb);
                ((uint4*)&bk)[1] = *(const uint4*)(bb + 16);
                sc[j] = __builtin_amdgcn_wmma_f32_16x16x32_f16(
                    false, aq, false, bk, (short)0, sc[j], false, false);
            }
        }

        // ---- online softmax per row (rows jj + 8*hs; 16-lane reductions) --
#pragma unroll
        for (int jj = 0; jj < 8; ++jj) {
            float mx = -1.0e30f;
#pragma unroll
            for (int j = 0; j < 4; ++j) {
                sc[j][jj] *= scale;
                mx = fmaxf(mx, sc[j][jj]);
            }
#pragma unroll
            for (int off = 1; off < 16; off <<= 1)
                mx = fmaxf(mx, __shfl_xor(mx, off, 32));
            const float mn   = fmaxf(mrow[jj], mx);
            const float corr = __expf(mrow[jj] - mn);
            float rs = 0.f;
#pragma unroll
            for (int j = 0; j < 4; ++j) {
                float p = __expf(sc[j][jj] - mn);
                sc[j][jj] = p;
                rs += p;
            }
#pragma unroll
            for (int off = 1; off < 16; off <<= 1)
                rs += __shfl_xor(rs, off, 32);
            lrow[jj] = lrow[jj] * corr + rs;
            mrow[jj] = mn;
#pragma unroll
            for (int j = 0; j < 4; ++j) o[j][jj] *= corr;
            // stage P into this wave's private LDS strip for re-fragmenting
            const int prow = wave * 16 + hs * 8 + jj;
#pragma unroll
            for (int j = 0; j < 4; ++j)
                Ps[prow * LDT + j * 16 + r] = (_Float16)sc[j][jj];
        }
        // wave-private LDS RAW hazard: cross-lane write -> fragment read
        asm volatile("s_wait_dscnt 0x0" ::: "memory");

        // ---- o += P @ V  (B operand = Vt, already [N=e][K=k]) ----
#pragma unroll
        for (int ks = 0; ks < 2; ++ks) {
            v16h ap;
            const _Float16* ab = Ps + (wave * 16 + r) * LDT + ks * 32 + hs * 8;
            ((uint4*)&ap)[0] = *(const uint4*)(ab);
            ((uint4*)&ap)[1] = *(const uint4*)(ab + 16);
#pragma unroll
            for (int j = 0; j < 4; ++j) {
                v16h bv;
                const _Float16* bb = Vt + (j * 16 + r) * LDT + ks * 32 + hs * 8;
                ((uint4*)&bv)[0] = *(const uint4*)(bb);
                ((uint4*)&bv)[1] = *(const uint4*)(bb + 16);
                o[j] = __builtin_amdgcn_wmma_f32_16x16x32_f16(
                    false, ap, false, bv, (short)0, o[j], false, false);
            }
        }
    }

    // ---- normalize and store Hcat[b, s, h*64 + e] ----
    float invl[8];
#pragma unroll
    for (int jj = 0; jj < 8; ++jj) invl[jj] = 1.0f / lrow[jj];
#pragma unroll
    for (int j = 0; j < 4; ++j) {
        const int e = j * 16 + r;
#pragma unroll
        for (int jj = 0; jj < 8; ++jj) {
            const int row = wave * 16 + hs * 8 + jj;
            Og[rowbase + (size_t)(q0 + row) * TD + e] = (_Float16)(o[j][jj] * invl[jj]);
        }
    }
}

// -------------------------------- launcher ---------------------------------
extern "C" void kernel_launch(void* const* d_in, const int* in_sizes, int n_in,
                              void* d_out, int out_size, void* d_ws, size_t ws_size,
                              hipStream_t stream) {
    const float* emb = (const float*)d_in[0];
    const float* Wq  = (const float*)d_in[1];
    const float* bq  = (const float*)d_in[2];
    const float* Wk  = (const float*)d_in[3];
    const float* bk  = (const float*)d_in[4];
    const float* Wv  = (const float*)d_in[5];
    const float* bv  = (const float*)d_in[6];
    const float* W1  = (const float*)d_in[7];
    const float* b1  = (const float*)d_in[8];
    const float* W2  = (const float*)d_in[9];
    const float* b2  = (const float*)d_in[10];
    float* out = (float*)d_out;

    // workspace carve-up (f16 units)
    _Float16* ws = (_Float16*)d_ws;
    const size_t nEmb = (size_t)TM * TD;          // 8192*1024
    const size_t nWqkv = (size_t)TD * TD;         // 1024*1024
    const size_t nW1 = (size_t)TFF * TD;
    const size_t nW2 = (size_t)TD * TFF;
    _Float16* emb16  = ws;                 ws += nEmb;
    _Float16* wq16   = ws;                 ws += nWqkv;
    _Float16* wk16   = ws;                 ws += nWqkv;
    _Float16* wv16   = ws;                 ws += nWqkv;
    _Float16* w1_16  = ws;                 ws += nW1;
    _Float16* w2_16  = ws;                 ws += nW2;
    _Float16* q16    = ws;                 ws += nEmb;
    _Float16* k16    = ws;                 ws += nEmb;
    _Float16* v16    = ws;                 ws += nEmb;
    _Float16* hcat16 = ws;                 ws += nEmb;
    _Float16* hid16  = ws;                 ws += (size_t)TM * TFF;

    auto cvt = [&](const float* src, _Float16* dst, size_t n) {
        cvt_f32_to_f16<<<dim3((unsigned)((n + 255) / 256)), dim3(256), 0, stream>>>(
            src, dst, (int)n);
    };
    cvt(emb, emb16, nEmb);
    cvt(Wq, wq16, nWqkv);
    cvt(Wk, wk16, nWqkv);
    cvt(Wv, wv16, nWqkv);
    cvt(W1, w1_16, nW1);
    cvt(W2, w2_16, nW2);

    // QKV projections: [8192,1024] = emb16 @ W^T + b   (f16 out)
    {
        dim3 grid(TD / 128, TM / 128), blk(256);
        gemm_wmma<false, _Float16><<<grid, blk, 0, stream>>>(emb16, wq16, bq, q16, TM, TD, TD);
        gemm_wmma<false, _Float16><<<grid, blk, 0, stream>>>(emb16, wk16, bk, k16, TM, TD, TD);
        gemm_wmma<false, _Float16><<<grid, blk, 0, stream>>>(emb16, wv16, bv, v16, TM, TD, TD);
    }

    // attention -> Hcat (f16, [B,S,H*HD])
    {
        dim3 grid(TS / 64, TB * TH), blk(128);
        attn_wmma<<<grid, blk, 0, stream>>>(q16, k16, v16, hcat16);
    }

    // FFN layer 1: relu(Hcat @ W1^T + b1)  -> hid16 [8192,4096]
    {
        dim3 grid(TFF / 128, TM / 128), blk(256);
        gemm_wmma<true, _Float16><<<grid, blk, 0, stream>>>(hcat16, w1_16, b1, hid16, TM, TFF, TD);
    }
    // FFN layer 2: hid @ W2^T + b2 -> d_out (f32, [8192,1024])
    {
        dim3 grid(TD / 128, TM / 128), blk(256);
        gemm_wmma<false, float><<<grid, blk, 0, stream>>>(hid16, w2_16, b2, out, TM, TD, TFF);
    }
}
